// DOGAT_12996571038312
// MI455X (gfx1250) — matrix operand order repaired and checked by reading, not compiled, not run on hardware
//
#include <hip/hip_runtime.h>
#include <hip/hip_bf16.h>
#include <stdint.h>

// ---------------------------------------------------------------------------
// Types for CDNA5 WMMA
// ---------------------------------------------------------------------------
typedef __bf16 bf16_t;
typedef __attribute__((ext_vector_type(16))) __bf16 v16bf;
typedef __attribute__((ext_vector_type(8)))  float  v8f;

#define GAT_H 10

// ---------------------------------------------------------------------------
// Small helpers
// ---------------------------------------------------------------------------
__global__ void k_zero_u32(uint32_t* p, long n) {
    long i = (long)blockIdx.x * blockDim.x + threadIdx.x;
    if (i < n) p[i] = 0u;
}

__global__ void k_f2bf(const float* __restrict__ in, bf16_t* __restrict__ out, long n) {
    long i = (long)blockIdx.x * blockDim.x + threadIdx.x;
    if (i < n) out[i] = (bf16_t)in[i];
}

__device__ __forceinline__ unsigned key_encode(float f) {
    unsigned u = __float_as_uint(f);
    return (u & 0x80000000u) ? ~u : (u | 0x80000000u);
}
__device__ __forceinline__ float key_decode(unsigned k) {
    unsigned u = (k & 0x80000000u) ? (k ^ 0x80000000u) : ~k;
    return __uint_as_float(u);
}

// ---------------------------------------------------------------------------
// bf16 WMMA GEMM:  C[M,N] = act(A[M,K] @ B[K,N] + bias)
// A,B row-major bf16, C float. act: 0=none 1=relu 2=elu
// Block: 256 threads (8 waves), tile 128x64, K-step 32.
// Fast (unguarded, vectorized) staging for interior blocks/full K tiles;
// guarded scalar staging only for edge blocks and the K tail.
// ---------------------------------------------------------------------------
#define BM 128
#define BN 64
#define BK 32
#define LDSP 40   // padded LDS row stride (elements) -> 80B rows, 16B aligned

struct Pack32 { uint4 a, b; };

__global__ __launch_bounds__(256) void k_gemm_bf16(
    const bf16_t* __restrict__ A, const bf16_t* __restrict__ Bm,
    const float* __restrict__ bias, float* __restrict__ C,
    int M, int N, int K, int act)
{
    __shared__ bf16_t As[BM * LDSP];
    __shared__ bf16_t Bs[BN * LDSP];

    const int tid  = threadIdx.x;
    const int lane = tid & 31;
    const int wave = tid >> 5;
    const int rowBlk = blockIdx.y * BM;
    const int colBlk = blockIdx.x * BN;
    const int wrow = (wave & 3) * 32;  // 4 waves along M
    const int wcol = (wave >> 2) * 32; // 2 waves along N

    // wave-uniform interiority (scalar branch, no per-lane exec churn)
    const bool interiorMN = (rowBlk + BM <= M) && (colBlk + BN <= N);

    v8f acc[2][2];
    #pragma unroll
    for (int i = 0; i < 2; i++)
        #pragma unroll
        for (int j = 0; j < 2; j++)
            acc[i][j] = (v8f)0.0f;

    for (int k0 = 0; k0 < K; k0 += BK) {
        if (interiorMN && (k0 + BK <= K)) {
            // ================= FAST PATH: straight-line staging =================
            // A tile: 128x32, 512 16B-chunks, 2 per thread
            #pragma unroll
            for (int c = tid; c < (BM * BK) / 8; c += 256) {
                int r   = c >> 2;          // 4 chunks per row
                int col = (c & 3) * 8;
                const uint32_t* s32 =
                    reinterpret_cast<const uint32_t*>(&A[(size_t)(rowBlk + r) * K + k0 + col]);
                uint32_t* d32 = reinterpret_cast<uint32_t*>(&As[r * LDSP + col]);
                d32[0] = s32[0]; d32[1] = s32[1]; d32[2] = s32[2]; d32[3] = s32[3];
            }
            // B tile transposed: Bs[n][k] = B[k0+k, colBlk+n]; 256 chunks, 1 per thread
            {
                int kr = tid >> 3;
                int nb = (tid & 7) * 8;
                const uint32_t* s32 =
                    reinterpret_cast<const uint32_t*>(&Bm[(size_t)(k0 + kr) * N + colBlk + nb]);
                union { uint32_t w[4]; bf16_t h[8]; } u;
                u.w[0] = s32[0]; u.w[1] = s32[1]; u.w[2] = s32[2]; u.w[3] = s32[3];
                #pragma unroll
                for (int j = 0; j < 8; j++)
                    Bs[(nb + j) * LDSP + kr] = u.h[j];
            }
            // prefetch next A K-tile
            if (k0 + BK < K) {
                int pr = rowBlk + (tid >> 1);
                if (pr < M) __builtin_prefetch(&A[(size_t)pr * K + k0 + BK], 0, 0);
            }
        } else {
            // ================= SLOW PATH: edge blocks / K tail =================
            #pragma unroll
            for (int c = tid; c < (BM * BK) / 8; c += 256) {
                int r   = c >> 2;
                int col = (c & 3) * 8;
                int gr  = rowBlk + r;
                int gc  = k0 + col;
                bf16_t* dst = &As[r * LDSP + col];
                if (gr < M && gc + 8 <= K) {
                    const uint32_t* s32 = reinterpret_cast<const uint32_t*>(&A[(size_t)gr * K + gc]);
                    uint32_t* d32 = reinterpret_cast<uint32_t*>(dst);
                    d32[0] = s32[0]; d32[1] = s32[1]; d32[2] = s32[2]; d32[3] = s32[3];
                } else {
                    #pragma unroll
                    for (int j = 0; j < 8; j++)
                        dst[j] = (gr < M && gc + j < K) ? A[(size_t)gr * K + gc + j] : (bf16_t)0.0f;
                }
            }
            {
                int kr = tid >> 3;
                int nb = (tid & 7) * 8;
                int gk = k0 + kr;
                #pragma unroll
                for (int j = 0; j < 8; j++) {
                    int gn = colBlk + nb + j;
                    bf16_t v = (gk < K && gn < N) ? Bm[(size_t)gk * N + gn] : (bf16_t)0.0f;
                    Bs[(nb + j) * LDSP + kr] = v;
                }
            }
        }
        __syncthreads();

        // ---- fragments from LDS (CDNA5 16-bit A(16x32)/B(32x16) layouts) ----
        const int kbA = (lane >> 4) * 8;   // A: lane<16 -> K 0..7 & 16..23 ; lane>=16 -> 8..15 & 24..31
        const int kbB = (lane >> 4) * 8;   // B: lane<16 -> K 0..7 & 16..23 ; lane>=16 -> 8..15 & 24..31
        const int sub = lane & 15;

        v16bf afrag[2], bfrag[2];
        #pragma unroll
        for (int mi = 0; mi < 2; mi++) {
            const uint4* p = reinterpret_cast<const uint4*>(&As[(wrow + mi * 16 + sub) * LDSP]);
            Pack32 t;
            t.a = p[kbA >> 3];            // halves 0..7
            t.b = p[(kbA >> 3) + 2];      // halves 8..15 (K+16)
            afrag[mi] = __builtin_bit_cast(v16bf, t);
        }
        #pragma unroll
        for (int ni = 0; ni < 2; ni++) {
            const uint4* p = reinterpret_cast<const uint4*>(&Bs[(wcol + ni * 16 + sub) * LDSP]);
            Pack32 t;
            t.a = p[kbB >> 3];
            t.b = p[(kbB >> 3) + 2];
            bfrag[ni] = __builtin_bit_cast(v16bf, t);
        }

        #pragma unroll
        for (int mi = 0; mi < 2; mi++)
            #pragma unroll
            for (int ni = 0; ni < 2; ni++)
                acc[mi][ni] = __builtin_amdgcn_wmma_f32_16x16x32_bf16(
                    false, afrag[mi], false, bfrag[ni],
                    (short)0, acc[mi][ni], false, false);

        __syncthreads();
    }

    // ---- epilogue: bias + activation + store (C/D layout: VGPR r -> M=r / r+8) ----
    const int sub   = lane & 15;
    const int mhalf = (lane >> 4) * 8;
    #pragma unroll
    for (int mi = 0; mi < 2; mi++) {
        #pragma unroll
        for (int ni = 0; ni < 2; ni++) {
            int coln = colBlk + wcol + ni * 16 + sub;
            #pragma unroll
            for (int r = 0; r < 8; r++) {
                int rowm = rowBlk + wrow + mi * 16 + mhalf + r;
                if (rowm < M && coln < N) {
                    float v = acc[mi][ni][r];
                    if (bias) v += bias[coln];
                    if (act == 1) v = fmaxf(v, 0.0f);
                    else if (act == 2) v = (v > 0.0f) ? v : (__expf(v) - 1.0f);
                    C[(size_t)rowm * N + coln] = v;
                }
            }
        }
    }
}

// ---------------------------------------------------------------------------
// Graph (GAT) kernels
// ---------------------------------------------------------------------------
__global__ void k_build_edges(const int* __restrict__ ei, int* __restrict__ src,
                              int* __restrict__ dst, int E, int N) {
    int i = blockIdx.x * blockDim.x + threadIdx.x;
    if (i < E) { src[i] = ei[i]; dst[i] = ei[E + i]; }
    else if (i < E + N) { src[i] = i - E; dst[i] = i - E; }
}

// per-node per-head attention scores
__global__ void k_gat_scores(const float* __restrict__ h, const float* __restrict__ a_s,
                             const float* __restrict__ a_d, float* __restrict__ ss,
                             float* __restrict__ sd, int N, int F) {
    int idx = blockIdx.x * blockDim.x + threadIdx.x;
    if (idx >= N * GAT_H) return;
    int n = idx / GAT_H, hd = idx % GAT_H;
    const float* hp = h + (size_t)n * GAT_H * F + (size_t)hd * F;
    float a = 0.0f, d = 0.0f;
    for (int f = 0; f < F; f++) { float v = hp[f]; a += v * a_s[hd * F + f]; d += v * a_d[hd * F + f]; }
    ss[idx] = a; sd[idx] = d;
}

__global__ void k_edge_max(const float* __restrict__ ss, const float* __restrict__ sd,
                           const int* __restrict__ src, const int* __restrict__ dst,
                           float* __restrict__ elk, unsigned* __restrict__ mkey, int Etot) {
    int e = blockIdx.x * blockDim.x + threadIdx.x;
    if (e >= Etot) return;
    int s = src[e], d = dst[e];
    #pragma unroll
    for (int hd = 0; hd < GAT_H; hd++) {
        float v = ss[s * GAT_H + hd] + sd[d * GAT_H + hd];
        v = (v > 0.0f) ? v : 0.2f * v;             // leaky_relu(0.2)
        elk[(size_t)e * GAT_H + hd] = v;
        atomicMax(&mkey[d * GAT_H + hd], key_encode(v));
    }
}

__global__ void k_edge_exp(const int* __restrict__ dst, const float* __restrict__ elk,
                           const unsigned* __restrict__ mkey, float* __restrict__ exv,
                           float* __restrict__ denom, int Etot) {
    int e = blockIdx.x * blockDim.x + threadIdx.x;
    if (e >= Etot) return;
    int d = dst[e];
    #pragma unroll
    for (int hd = 0; hd < GAT_H; hd++) {
        float m = key_decode(mkey[d * GAT_H + hd]);
        float x = __expf(elk[(size_t)e * GAT_H + hd] - m);
        exv[(size_t)e * GAT_H + hd] = x;
        atomicAdd(&denom[d * GAT_H + hd], x);
    }
}

// out[dst, hd, f] += h[src, hd, f] * ex/denom ; one block per edge
__global__ void k_gat_aggregate(const float* __restrict__ h, const int* __restrict__ src,
                                const int* __restrict__ dst, const float* __restrict__ exv,
                                const float* __restrict__ denom, float* __restrict__ out,
                                int Etot, int F) {
    int e = blockIdx.x;
    if (e >= Etot) return;
    int s = src[e], d = dst[e];
    int HF = GAT_H * F;
    for (int j = threadIdx.x; j < HF; j += blockDim.x) {
        int hd = j / F;
        float alpha = exv[(size_t)e * GAT_H + hd] / denom[d * GAT_H + hd];
        atomicAdd(&out[(size_t)d * HF + j], h[(size_t)s * HF + j] * alpha);
    }
}

// concat heads: elu(agg + b1) -> bf16 for next GEMM
__global__ void k_gat1_finalize(const float* __restrict__ agg, const float* __restrict__ b1,
                                bf16_t* __restrict__ out_bf, long n, int C) {
    long i = (long)blockIdx.x * blockDim.x + threadIdx.x;
    if (i >= n) return;
    float v = agg[i] + b1[i % C];
    v = (v > 0.0f) ? v : (__expf(v) - 1.0f);
    out_bf[i] = (bf16_t)v;
}

// head-mean: relu(mean_h agg + b2) -> float [N,780]
__global__ void k_gat2_finalize(const float* __restrict__ agg, const float* __restrict__ b2,
                                float* __restrict__ out, int N, int F) {
    long i = (long)blockIdx.x * blockDim.x + threadIdx.x;
    if (i >= (long)N * F) return;
    int n = (int)(i / F), f = (int)(i % F);
    float s = 0.0f;
    #pragma unroll
    for (int hd = 0; hd < GAT_H; hd++) s += agg[(size_t)n * GAT_H * F + (size_t)hd * F + f];
    float v = s * (1.0f / GAT_H) + b2[f];
    out[i] = fmaxf(v, 0.0f);
}

// ---------------------------------------------------------------------------
// Pooling / misc kernels
// ---------------------------------------------------------------------------
__global__ void k_count_nodes(const int* __restrict__ batch, float* __restrict__ cnt, int N) {
    int n = blockIdx.x * blockDim.x + threadIdx.x;
    if (n < N) atomicAdd(&cnt[batch[n]], 1.0f);
}

__global__ void k_pool_pass(const float* __restrict__ h, const int* __restrict__ batch,
                            unsigned* __restrict__ gmaxb, float* __restrict__ gsum, int N) {
    long i = (long)blockIdx.x * blockDim.x + threadIdx.x;
    if (i >= (long)N * 780) return;
    int n = (int)(i / 780), f = (int)(i % 780);
    float v = h[i];
    int b = batch[n];
    atomicMax(&gmaxb[b * 780 + f], __float_as_uint(v));  // v >= 0 after relu
    atomicAdd(&gsum[b * 780 + f], v);
}

__global__ void k_build_gfeat(const unsigned* __restrict__ gmaxb, const float* __restrict__ gsum,
                              const float* __restrict__ cnt, bf16_t* __restrict__ gfeat, int B) {
    int i = blockIdx.x * blockDim.x + threadIdx.x;
    if (i >= B * 1560) return;
    int b = i / 1560, j = i % 1560;
    float v = (j < 780) ? __uint_as_float(gmaxb[b * 780 + j])
                        : gsum[b * 780 + (j - 780)] / fmaxf(cnt[b], 1.0f);
    gfeat[i] = (bf16_t)v;
}

__global__ void k_embed_gather(const int* __restrict__ target, const float* __restrict__ emb,
                               float* __restrict__ e, int B) {
    long i = (long)blockIdx.x * blockDim.x + threadIdx.x;
    if (i >= (long)B * 1000 * 128) return;
    int b = (int)(i / 128000);
    int r = (int)(i % 128000);
    int ch = r / 128, l = r % 128;
    e[i] = emb[(size_t)target[b * 1000 + ch] * 128 + l];
}

// Conv1d: channels=1000 (seq tokens), kernel=8 over the 128 embed axis -> [B,32,121]
__global__ void k_conv1d(const float* __restrict__ e, const float* __restrict__ Wc,
                         const float* __restrict__ bc, float* __restrict__ out, int B) {
    int i = blockIdx.x * blockDim.x + threadIdx.x;
    if (i >= B * 32 * 121) return;
    int b = i / (32 * 121);
    int o = (i / 121) % 32;
    int p = i % 121;
    const float* eb = e + (size_t)b * 1000 * 128;
    const float* w  = Wc + (size_t)o * 8000;
    float acc = 0.0f;
    for (int c = 0; c < 1000; c++) {
        const float* er = eb + (size_t)c * 128 + p;
        const float* wr = w + c * 8;
        #pragma unroll
        for (int k = 0; k < 8; k++) acc += er[k] * wr[k];
    }
    out[(size_t)b * 3872 + o * 121 + p] = acc + bc[o];
}

__global__ void k_build_xc(const float* __restrict__ g, const float* __restrict__ xt,
                           bf16_t* __restrict__ xc, int B) {
    int i = blockIdx.x * blockDim.x + threadIdx.x;
    if (i >= B * 256) return;
    int b = i / 256, j = i % 256;
    float v = (j < 128) ? g[b * 128 + j] : xt[b * 128 + (j - 128)];
    xc[i] = (bf16_t)v;
}

__global__ void k_final_head(const float* __restrict__ f2, const float* __restrict__ Wo,
                             const float* __restrict__ bo, float* __restrict__ out, int B) {
    int b = blockIdx.x * blockDim.x + threadIdx.x;
    if (b >= B) return;
    float acc = bo[0];
    for (int k = 0; k < 512; k++) acc += f2[(size_t)b * 512 + k] * Wo[k];
    out[b] = acc;
}

// ---------------------------------------------------------------------------
// Host-side orchestration
// ---------------------------------------------------------------------------
static inline dim3 grid1(long n, int blk = 256) { return dim3((unsigned)((n + blk - 1) / blk)); }

extern "C" void kernel_launch(void* const* d_in, const int* in_sizes, int n_in,
                              void* d_out, int out_size, void* d_ws, size_t ws_size,
                              hipStream_t stream) {
    // ---- inputs (setup_inputs order) ----
    const float* x    = (const float*)d_in[0];
    const int*   ei   = (const int*)  d_in[1];
    const int*   batch= (const int*)  d_in[2];
    const int*   tgt  = (const int*)  d_in[3];
    const float* W1   = (const float*)d_in[4];
    const float* a1s  = (const float*)d_in[5];
    const float* a1d  = (const float*)d_in[6];
    const float* b1   = (const float*)d_in[7];
    const float* W2   = (const float*)d_in[8];
    const float* a2s  = (const float*)d_in[9];
    const float* a2d  = (const float*)d_in[10];
    const float* b2   = (const float*)d_in[11];
    const float* Wg1  = (const float*)d_in[12];
    const float* bg1  = (const float*)d_in[13];
    const float* Wg2  = (const float*)d_in[14];
    const float* bg2  = (const float*)d_in[15];
    const float* emb  = (const float*)d_in[16];
    const float* Wc   = (const float*)d_in[17];
    const float* bc   = (const float*)d_in[18];
    const float* Wxt  = (const float*)d_in[19];
    const float* bxt  = (const float*)d_in[20];
    const float* Wf1  = (const float*)d_in[21];
    const float* bf1  = (const float*)d_in[22];
    const float* Wf2  = (const float*)d_in[23];
    const float* bf2  = (const float*)d_in[24];
    const float* Wo   = (const float*)d_in[25];
    const float* bo   = (const float*)d_in[26];

    const int N = in_sizes[0] / 78;
    const int E = in_sizes[1] / 2;
    const int B = in_sizes[3] / 1000;
    const int Etot = E + N;

    // ---- workspace bump allocator ----
    size_t off = 0;
    auto alloc = [&](size_t bytes) -> void* {
        void* p = (char*)d_ws + off;
        off = (off + bytes + 255) & ~(size_t)255;
        return p;
    };
    bf16_t* x_bf   = (bf16_t*)alloc((size_t)N * 78 * 2);
    bf16_t* W1_bf  = (bf16_t*)alloc((size_t)78 * 780 * 2);
    bf16_t* W2_bf  = (bf16_t*)alloc((size_t)780 * 7800 * 2);
    bf16_t* Wg1_bf = (bf16_t*)alloc((size_t)1560 * 1500 * 2);
    bf16_t* Wg2_bf = (bf16_t*)alloc((size_t)1500 * 128 * 2);
    bf16_t* Wxt_bf = (bf16_t*)alloc((size_t)3872 * 128 * 2);
    bf16_t* Wf1_bf = (bf16_t*)alloc((size_t)256 * 1024 * 2);
    bf16_t* Wf2_bf = (bf16_t*)alloc((size_t)1024 * 512 * 2);
    int*    srcF   = (int*)   alloc((size_t)Etot * 4);
    int*    dstF   = (int*)   alloc((size_t)Etot * 4);
    float*  h1raw  = (float*) alloc((size_t)N * 780 * 4);
    float*  ss     = (float*) alloc((size_t)N * GAT_H * 4);
    float*  sd     = (float*) alloc((size_t)N * GAT_H * 4);
    float*  elk    = (float*) alloc((size_t)Etot * GAT_H * 4);
    float*  exv    = (float*) alloc((size_t)Etot * GAT_H * 4);
    unsigned* mkey = (unsigned*)alloc((size_t)N * GAT_H * 4);
    float*  denom  = (float*) alloc((size_t)N * GAT_H * 4);
    float*  agg1   = (float*) alloc((size_t)N * 780 * 4);
    bf16_t* h1bf   = (bf16_t*)alloc((size_t)N * 780 * 2);
    float*  h2raw  = (float*) alloc((size_t)N * 7800 * 4);
    float*  agg2   = (float*) alloc((size_t)N * 7800 * 4);
    float*  h2act  = (float*) alloc((size_t)N * 780 * 4);
    unsigned* gmaxb= (unsigned*)alloc((size_t)B * 780 * 4);
    float*  gsum   = (float*) alloc((size_t)B * 780 * 4);
    float*  cnt    = (float*) alloc((size_t)B * 4);
    bf16_t* gfeatb = (bf16_t*)alloc((size_t)B * 1560 * 2);
    float*  g1     = (float*) alloc((size_t)B * 1500 * 4);
    bf16_t* g1bf   = (bf16_t*)alloc((size_t)B * 1500 * 2);
    float*  gfeat2 = (float*) alloc((size_t)B * 128 * 4);
    float*  eemb   = (float*) alloc((size_t)B * 1000 * 128 * 4);
    float*  convf  = (float*) alloc((size_t)B * 3872 * 4);
    bf16_t* convbf = (bf16_t*)alloc((size_t)B * 3872 * 2);
    float*  xt     = (float*) alloc((size_t)B * 128 * 4);
    bf16_t* xcbf   = (bf16_t*)alloc((size_t)B * 256 * 2);
    float*  f1     = (float*) alloc((size_t)B * 1024 * 4);
    bf16_t* f1bf   = (bf16_t*)alloc((size_t)B * 1024 * 2);
    float*  f2     = (float*) alloc((size_t)B * 512 * 4);

    // ---- convert weights + x to bf16 ----
    k_f2bf<<<grid1((long)N * 78), 256, 0, stream>>>(x, x_bf, (long)N * 78);
    k_f2bf<<<grid1(78L * 780), 256, 0, stream>>>(W1, W1_bf, 78L * 780);
    k_f2bf<<<grid1(780L * 7800), 256, 0, stream>>>(W2, W2_bf, 780L * 7800);
    k_f2bf<<<grid1(1560L * 1500), 256, 0, stream>>>(Wg1, Wg1_bf, 1560L * 1500);
    k_f2bf<<<grid1(1500L * 128), 256, 0, stream>>>(Wg2, Wg2_bf, 1500L * 128);
    k_f2bf<<<grid1(3872L * 128), 256, 0, stream>>>(Wxt, Wxt_bf, 3872L * 128);
    k_f2bf<<<grid1(256L * 1024), 256, 0, stream>>>(Wf1, Wf1_bf, 256L * 1024);
    k_f2bf<<<grid1(1024L * 512), 256, 0, stream>>>(Wf2, Wf2_bf, 1024L * 512);

    // ---- edges with self loops ----
    k_build_edges<<<grid1(Etot), 256, 0, stream>>>(ei, srcF, dstF, E, N);

    // ================= GAT layer 1 (78 -> 10x78, concat) =================
    {
        dim3 g((780 + BN - 1) / BN, (N + BM - 1) / BM);
        k_gemm_bf16<<<g, 256, 0, stream>>>(x_bf, W1_bf, nullptr, h1raw, N, 780, 78, 0);
    }
    k_gat_scores<<<grid1((long)N * GAT_H), 256, 0, stream>>>(h1raw, a1s, a1d, ss, sd, N, 78);
    k_zero_u32<<<grid1((long)N * GAT_H), 256, 0, stream>>>(mkey, (long)N * GAT_H);
    k_zero_u32<<<grid1((long)N * GAT_H), 256, 0, stream>>>((uint32_t*)denom, (long)N * GAT_H);
    k_zero_u32<<<grid1((long)N * 780), 256, 0, stream>>>((uint32_t*)agg1, (long)N * 780);
    k_edge_max<<<grid1(Etot), 256, 0, stream>>>(ss, sd, srcF, dstF, elk, mkey, Etot);
    k_edge_exp<<<grid1(Etot), 256, 0, stream>>>(dstF, elk, mkey, exv, denom, Etot);
    k_gat_aggregate<<<dim3(Etot), 256, 0, stream>>>(h1raw, srcF, dstF, exv, denom, agg1, Etot, 78);
    k_gat1_finalize<<<grid1((long)N * 780), 256, 0, stream>>>(agg1, b1, h1bf, (long)N * 780, 780);

    // ================= GAT layer 2 (780 -> 10x780, head-mean) =================
    {
        dim3 g((7800 + BN - 1) / BN, (N + BM - 1) / BM);
        k_gemm_bf16<<<g, 256, 0, stream>>>(h1bf, W2_bf, nullptr, h2raw, N, 7800, 780, 0);
    }
    k_gat_scores<<<grid1((long)N * GAT_H), 256, 0, stream>>>(h2raw, a2s, a2d, ss, sd, N, 780);
    k_zero_u32<<<grid1((long)N * GAT_H), 256, 0, stream>>>(mkey, (long)N * GAT_H);
    k_zero_u32<<<grid1((long)N * GAT_H), 256, 0, stream>>>((uint32_t*)denom, (long)N * GAT_H);
    k_zero_u32<<<grid1((long)N * 7800), 256, 0, stream>>>((uint32_t*)agg2, (long)N * 7800);
    k_edge_max<<<grid1(Etot), 256, 0, stream>>>(ss, sd, srcF, dstF, elk, mkey, Etot);
    k_edge_exp<<<grid1(Etot), 256, 0, stream>>>(dstF, elk, mkey, exv, denom, Etot);
    k_gat_aggregate<<<dim3(Etot), 256, 0, stream>>>(h2raw, srcF, dstF, exv, denom, agg2, Etot, 780);
    k_gat2_finalize<<<grid1((long)N * 780), 256, 0, stream>>>(agg2, b2, h2act, N, 780);

    // ================= batch pooling (max & mean) =================
    k_zero_u32<<<grid1((long)B * 780), 256, 0, stream>>>(gmaxb, (long)B * 780);
    k_zero_u32<<<grid1((long)B * 780), 256, 0, stream>>>((uint32_t*)gsum, (long)B * 780);
    k_zero_u32<<<grid1(B), 64, 0, stream>>>((uint32_t*)cnt, B);
    k_count_nodes<<<grid1(N), 256, 0, stream>>>(batch, cnt, N);
    k_pool_pass<<<grid1((long)N * 780), 256, 0, stream>>>(h2act, batch, gmaxb, gsum, N);
    k_build_gfeat<<<grid1((long)B * 1560), 256, 0, stream>>>(gmaxb, gsum, cnt, gfeatb, B);

    // graph MLP: 1560 -> 1500 (relu) -> 128 (relu)
    {
        dim3 g((1500 + BN - 1) / BN, (B + BM - 1) / BM);
        k_gemm_bf16<<<g, 256, 0, stream>>>(gfeatb, Wg1_bf, bg1, g1, B, 1500, 1560, 1);
    }
    k_f2bf<<<grid1((long)B * 1500), 256, 0, stream>>>(g1, g1bf, (long)B * 1500);
    {
        dim3 g((128 + BN - 1) / BN, (B + BM - 1) / BM);
        k_gemm_bf16<<<g, 256, 0, stream>>>(g1bf, Wg2_bf, bg2, gfeat2, B, 128, 1500, 1);
    }

    // ================= protein branch =================
    k_embed_gather<<<grid1((long)B * 1000 * 128), 256, 0, stream>>>(tgt, emb, eemb, B);
    k_conv1d<<<grid1((long)B * 32 * 121), 256, 0, stream>>>(eemb, Wc, bc, convf, B);
    k_f2bf<<<grid1((long)B * 3872), 256, 0, stream>>>(convf, convbf, (long)B * 3872);
    {
        dim3 g((128 + BN - 1) / BN, (B + BM - 1) / BM);
        k_gemm_bf16<<<g, 256, 0, stream>>>(convbf, Wxt_bf, bxt, xt, B, 128, 3872, 0);
    }

    // ================= fusion head =================
    k_build_xc<<<grid1((long)B * 256), 256, 0, stream>>>(gfeat2, xt, xcbf, B);
    {
        dim3 g((1024 + BN - 1) / BN, (B + BM - 1) / BM);
        k_gemm_bf16<<<g, 256, 0, stream>>>(xcbf, Wf1_bf, bf1, f1, B, 1024, 256, 1);
    }
    k_f2bf<<<grid1((long)B * 1024), 256, 0, stream>>>(f1, f1bf, (long)B * 1024);
    {
        dim3 g((512 + BN - 1) / BN, (B + BM - 1) / BM);
        k_gemm_bf16<<<g, 256, 0, stream>>>(f1bf, Wf2_bf, bf2, f2, B, 512, 1024, 1);
    }
    k_final_head<<<grid1(B), 64, 0, stream>>>(f2, Wo, bo, (float*)d_out, B);
}
